// GNN12_27410481283381
// MI455X (gfx1250) — compile-verified, matrix operand order checked by reading, not run on hardware
//
#include <hip/hip_runtime.h>
#include <hip/hip_bf16.h>
#include <hip/hip_fp16.h>

typedef __attribute__((ext_vector_type(16))) _Float16 v16h;
typedef __attribute__((ext_vector_type(8)))  _Float16 v8h;
typedef __attribute__((ext_vector_type(8)))  float    v8f;

#define GB   128      // batches
#define GN   2048     // nodes
#define GK   20       // neighbors
#define FIN  11       // input feats
#define FOUT 256      // output feats
#define FKP  32       // padded K for WMMA (22 used: [x | nb_mean])
#define NODES_PER_WG 128
#define TILES 16      // FOUT / 16
#define WFRAG_HALFS (TILES * 32 * 16)          // 8192 halfs = 16 KiB

// workspace layout (bytes)
#define WS_WFRAG_OFF 0                         // 16 KiB packed B fragments
#define WS_ACC_OFF   16384                     // float [2][GB][FOUT]
#define WS_ESUM_OFF  (16384 + 2*GB*FOUT*4)     // float [2][GB]
#define WS_ZERO_FLOATS (2*GB*FOUT + 2*GB)      // acc + esum contiguous

// ---------------------------------------------------------------- zero accum
__global__ void k_zero(float* __restrict__ p, int n) {
    int i = blockIdx.x * blockDim.x + threadIdx.x;
    if (i < n) p[i] = 0.0f;
}

// ------------------------------------------------- pack W=[Wc;Wn] (22x256) to
// f16 B-fragment layout: lane<16 holds col N=lane, K={0..7,16..23};
// lane>=16 holds col N=lane-16, K={8..15,24..31}. 16 contiguous halfs/lane.
__global__ void k_pack_w(const float* __restrict__ Wc,
                         const float* __restrict__ Wn,
                         _Float16* __restrict__ wfrag) {
    int t    = blockIdx.x;        // tile 0..15 (output cols t*16..t*16+15)
    int lane = threadIdx.x;       // 0..31
    int n    = lane & 15;
    int hi   = lane >> 4;
    _Float16* dst = wfrag + (t * 32 + lane) * 16;
    int col = t * 16 + n;
#pragma unroll
    for (int i = 0; i < 16; ++i) {
        int k = (i < 8) ? (hi * 8 + i) : (16 + hi * 8 + (i - 8));
        float v = 0.0f;
        if (k < FIN)            v = Wc[k * FOUT + col];
        else if (k < 2 * FIN)   v = Wn[(k - FIN) * FOUT + col];
        dst[i] = (_Float16)v;
    }
}

// ---------------------------------------------------------------- fused main
// grid = (GN/128, GB, 2 streams), block = 256 (8 waves, 16 nodes/wave).
__global__ __launch_bounds__(256)
void k_main(const float* __restrict__ x0, const int* __restrict__ nbr0,
            const float* __restrict__ watt0,
            const float* __restrict__ x1, const int* __restrict__ nbr1,
            const float* __restrict__ watt1,
            const float* __restrict__ bconv,
            const _Float16* __restrict__ wfrag,
            float* __restrict__ acc, float* __restrict__ esum) {
    __shared__ _Float16 s_feat[NODES_PER_WG][FKP];   // 8 KiB
    __shared__ _Float16 s_wfrag[WFRAG_HALFS];        // 16 KiB (B fragments)
    __shared__ float    s_acc[FOUT];                 // 1 KiB
    __shared__ float    s_esum;

    const int tid    = threadIdx.x;
    const int b      = blockIdx.y;
    const int strm   = blockIdx.z;
    const int nbase  = blockIdx.x * NODES_PER_WG;

    const float* x    = strm ? x1 : x0;
    const int*   nbr  = strm ? nbr1 : nbr0;
    const float* watt = strm ? watt1 : watt0;

    if (tid < FOUT) s_acc[tid] = 0.0f;
    if (tid == 0)   s_esum = 0.0f;

    // ---- async DMA: stage packed B fragments (16 KiB) into LDS once per WG,
    // overlapped with the gather-heavy phase 1. GVS mode: SGPR64 base +
    // VGPR32 byte offset; VDST = LDS byte address (low 32b of flat shared ptr).
    {
        unsigned long long sbase = (unsigned long long)wfrag;
#pragma unroll
        for (int it = 0; it < 4; ++it) {               // 256 thr * 16 B * 4 = 16 KiB
            unsigned int goff = (unsigned int)(tid * 16 + it * 4096);
            unsigned int ldsa =
                (unsigned int)(uintptr_t)((char*)s_wfrag + tid * 16 + it * 4096);
            asm volatile(
                "global_load_async_to_lds_b128 %0, %1, %2"
                :: "v"(ldsa), "v"(goff), "s"(sbase)
                : "memory");
        }
    }

    // ---- phase 1: build padded f16 features in LDS: [x(11) | mean_nb(11) | 0(10)]
    {
        int n    = tid >> 1;                 // local node 0..127
        int s    = tid & 1;                  // neighbor-half split
        int node = nbase + n;
        const float* xrow = x   + ((size_t)b * GN + node) * FIN;
        const int*   nrow = nbr + ((size_t)b * GN + node) * GK;
        float nsum[FIN];
#pragma unroll
        for (int f = 0; f < FIN; ++f) nsum[f] = 0.0f;
        for (int k = 0; k < GK / 2; ++k) {
            int idx = nrow[s * (GK / 2) + k];
            const float* nr = x + ((size_t)b * GN + idx) * FIN;
#pragma unroll
            for (int f = 0; f < FIN; ++f) nsum[f] += nr[f];
        }
#pragma unroll
        for (int f = 0; f < FIN; ++f) {
            float v = nsum[f];
            v += __shfl_xor(v, 1);           // combine the two half-gathers
            if (s == 0) s_feat[n][FIN + f] = (_Float16)(v * (1.0f / GK));
        }
        if (s == 0) {
#pragma unroll
            for (int f = 0; f < FIN; ++f) s_feat[n][f] = (_Float16)xrow[f];
#pragma unroll
            for (int f = 2 * FIN; f < FKP; ++f) s_feat[n][f] = (_Float16)0.0f;
        }
    }
    asm volatile("s_wait_asynccnt 0" ::: "memory");    // DMA complete (this wave)
    __syncthreads();                                   // all waves' writes visible

    // ---- phase 2: per-wave WMMA 16 nodes x 256 outputs, fused attention
    const int wave = tid >> 5;
    const int lane = tid & 31;
    const int grp  = lane >> 4;              // half-wave group
    const int lcol = lane & 15;
    const int m    = wave * 16 + lcol;       // A row for this lane (both groups)

    v16h afrag;
    {
        int o = grp * 8;                     // lanes<16: K 0..7,16..23 ; >=16: 8..15,24..31
        v8h lo = *(const v8h*)&s_feat[m][o];
        v8h hi = *(const v8h*)&s_feat[m][o + 16];
#pragma unroll
        for (int i = 0; i < 8; ++i) { afrag[i] = lo[i]; afrag[i + 8] = hi[i]; }
    }

    v8f   c[TILES];
    float dotv[8];
#pragma unroll
    for (int r = 0; r < 8; ++r) dotv[r] = 0.0f;

#pragma unroll
    for (int t = 0; t < TILES; ++t) {
        const v8h* wp = (const v8h*)(s_wfrag + (t * 32 + lane) * 16);
        v8h blo = wp[0], bhi = wp[1];
        v16h bfrag;
#pragma unroll
        for (int i = 0; i < 8; ++i) { bfrag[i] = blo[i]; bfrag[i + 8] = bhi[i]; }

        v8f cz = {};
        c[t] = __builtin_amdgcn_wmma_f32_16x16x32_f16(
            /*neg_a=*/false, afrag, /*neg_b=*/false, bfrag,
            /*c_mod=*/(short)0, cz, /*reuse_a=*/false, /*reuse_b=*/false);

        int   col  = t * 16 + lcol;
        float bias = bconv[col];
        float w    = watt[col];
#pragma unroll
        for (int r = 0; r < 8; ++r) {
            float v = c[t][r] + bias;
            v = fmaxf(v, 0.0f);              // ReLU
            c[t][r] = v;
            float p = v * w;                 // attention-dot partial
            p += __shfl_xor(p, 1);
            p += __shfl_xor(p, 2);
            p += __shfl_xor(p, 4);
            p += __shfl_xor(p, 8);           // 16-lane reduce: row dot broadcast
            dotv[r] += p;
        }
    }

    // scores: e = exp(tanh(dot)) — tanh bounded, safe without max-subtract
    float e[8], te = 0.0f;
#pragma unroll
    for (int r = 0; r < 8; ++r) { e[r] = __expf(tanhf(dotv[r])); te += e[r]; }

    // weighted column pooling into LDS, then one global atomic per column
#pragma unroll
    for (int t = 0; t < TILES; ++t) {
        float csum = 0.0f;
#pragma unroll
        for (int r = 0; r < 8; ++r) csum += e[r] * c[t][r];
        csum += __shfl_xor(csum, 16);        // combine row groups 0..7 / 8..15
        if (grp == 0) atomicAdd(&s_acc[t * 16 + lcol], csum);
    }
    te += __shfl_xor(te, 16);
    if (lane == 0) atomicAdd(&s_esum, te);

    __syncthreads();
    float* accst = acc + ((size_t)strm * GB + b) * FOUT;
    if (tid < FOUT) atomicAdd(&accst[tid], s_acc[tid]);
    if (tid == 0)   atomicAdd(&esum[strm * GB + b], s_esum);
}

// --------------------------------------------------------------- finalize
__global__ void k_final(const float* __restrict__ acc,
                        const float* __restrict__ esum,
                        const float* __restrict__ Wd,
                        const float* __restrict__ bd,
                        float* __restrict__ out) {
    __shared__ float s_part[8];
    int b = blockIdx.x, tid = threadIdx.x;   // 256 threads = FOUT
    float ri = acc[(size_t)b * FOUT + tid] / esum[b];
    float rn = acc[((size_t)GB + b) * FOUT + tid] / esum[GB + b];
    float v  = ri * Wd[tid] + rn * Wd[FOUT + tid];
#pragma unroll
    for (int msk = 16; msk >= 1; msk >>= 1) v += __shfl_xor(v, msk);
    if ((tid & 31) == 0) s_part[tid >> 5] = v;
    __syncthreads();
    if (tid == 0) {
        float s = 0.0f;
#pragma unroll
        for (int i = 0; i < 8; ++i) s += s_part[i];
        out[b] = s + bd[0];
    }
}

// ---------------------------------------------------------------- launcher
extern "C" void kernel_launch(void* const* d_in, const int* in_sizes, int n_in,
                              void* d_out, int out_size, void* d_ws, size_t ws_size,
                              hipStream_t stream) {
    const float* x_int    = (const float*)d_in[0];
    const int*   nbr_int  = (const int*)d_in[1];
    const float* x_nh     = (const float*)d_in[2];
    const int*   nbr_nh   = (const int*)d_in[3];
    const float* Wc       = (const float*)d_in[4];
    const float* Wn       = (const float*)d_in[5];
    const float* bconv    = (const float*)d_in[6];
    const float* watt_int = (const float*)d_in[7];
    const float* watt_nh  = (const float*)d_in[8];
    const float* Wd       = (const float*)d_in[9];
    const float* bd       = (const float*)d_in[10];
    float* out = (float*)d_out;

    _Float16* wfrag = (_Float16*)((char*)d_ws + WS_WFRAG_OFF);
    float*    acc   = (float*)((char*)d_ws + WS_ACC_OFF);
    float*    esumv = (float*)((char*)d_ws + WS_ESUM_OFF);

    int nz = WS_ZERO_FLOATS;
    k_zero<<<(nz + 255) / 256, 256, 0, stream>>>(acc, nz);
    k_pack_w<<<TILES, 32, 0, stream>>>(Wc, Wn, wfrag);

    dim3 grid(GN / NODES_PER_WG, GB, 2);
    k_main<<<grid, 256, 0, stream>>>(x_int, nbr_int, watt_int,
                                     x_nh, nbr_nh, watt_nh,
                                     bconv, wfrag, acc, esumv);

    k_final<<<GB, FOUT, 0, stream>>>(acc, esumv, Wd, bd, out);
}
